// MHA_62466004353111
// MI455X (gfx1250) — matrix operand order repaired and checked by reading, not compiled, not run on hardware
//
#include <hip/hip_runtime.h>

// MHA forward for B=4, T=2048, D=1024, H=16, hd=64 (fp32 in/out).
// bf16 WMMA (v_wmma_f32_16x16x32_bf16) for all matmuls, fp32 accumulation,
// flash-style causal softmax. Weights pre-transposed so all LDS staging is
// b128; V produced transposed [B,H,hd,T] so attention stages V with b128 too.

#define DMODEL 1024
#define NHEAD  16
#define HDIM   64
#define SEQ    2048
#define BATCH  4
#define MROWS  (BATCH * SEQ)   // 8192

typedef __attribute__((ext_vector_type(16))) __bf16 bf16x16;
typedef __attribute__((ext_vector_type(8)))  __bf16 bf16x8;
typedef __attribute__((ext_vector_type(4)))  __bf16 bf16x4;
typedef __attribute__((ext_vector_type(8)))  float  f32x8;

__device__ __forceinline__ f32x8 zero8() {
  f32x8 z = {0.f, 0.f, 0.f, 0.f, 0.f, 0.f, 0.f, 0.f};
  return z;
}

// Build a 16-element bf16 fragment from two aligned 8-element (b128) loads.
__device__ __forceinline__ bf16x16 ld_frag(const __bf16* lo, const __bf16* hi) {
  bf16x8 a = *(const bf16x8*)lo;
  bf16x8 b = *(const bf16x8*)hi;
  bf16x16 r;
#pragma unroll
  for (int i = 0; i < 8; ++i) { r[i] = a[i]; r[i + 8] = b[i]; }
  return r;
}

__device__ __forceinline__ f32x8 wmma_bf16(bf16x16 a, bf16x16 b, f32x8 c) {
  // (neg_a, A, neg_b, B, c_mod, C, reuse_a, reuse_b)
  return __builtin_amdgcn_wmma_f32_16x16x32_bf16(false, a, false, b, (short)0, c,
                                                 false, false);
}

// ---------------------------------------------------------------------------
// fp32 -> bf16 conversion, 4 elements / thread (for x)
// ---------------------------------------------------------------------------
__global__ __launch_bounds__(256) void cvt_f32_bf16(const float* __restrict__ in,
                                                    __bf16* __restrict__ out,
                                                    int n4) {
  int i = blockIdx.x * blockDim.x + threadIdx.x;
  if (i >= n4) return;
  float4 v = ((const float4*)in)[i];
  bf16x4 r;
  r[0] = (__bf16)v.x; r[1] = (__bf16)v.y; r[2] = (__bf16)v.z; r[3] = (__bf16)v.w;
  ((bf16x4*)out)[i] = r;
}

// ---------------------------------------------------------------------------
// fp32 [K][N] -> bf16 transposed [N][K] (weights; 32x32 LDS tile transpose)
// ---------------------------------------------------------------------------
__global__ __launch_bounds__(256) void cvt_transpose_bf16(const float* __restrict__ in,
                                                          __bf16* __restrict__ out,
                                                          int K, int N) {
  __shared__ float tile[32][33];
  const int k0 = blockIdx.x * 32;
  const int n0 = blockIdx.y * 32;
  const int tx = threadIdx.x & 31;
  const int ty = threadIdx.x >> 5;  // 0..7
#pragma unroll
  for (int i = 0; i < 4; ++i)
    tile[ty + i * 8][tx] = in[(size_t)(k0 + ty + i * 8) * N + n0 + tx];
  __syncthreads();
#pragma unroll
  for (int i = 0; i < 4; ++i)
    out[(size_t)(n0 + ty + i * 8) * K + k0 + tx] = (__bf16)tile[tx][ty + i * 8];
}

// ---------------------------------------------------------------------------
// bf16 WMMA GEMM:  C[M,N] = A[M,K] * Wt[N,K]^T + bias[N]
//   MODE 0: bf16 out, permuted [B,T,H*hd] -> [B,H,T,hd]      (Q,K projections)
//   MODE 1: fp32 out, plain row-major [M,N]                  (output projection)
//   MODE 2: bf16 out, transposed   -> [B,H,hd,T], b128 packed (V projection)
// Block: 256 threads = 8 waves; tile 128x128, K-step 32, double-buffered LDS.
// Wave (wy in 0..3, wx in 0..1) computes 32x64 = 2x4 WMMA tiles.
// ---------------------------------------------------------------------------
template <int MODE>
__global__ __launch_bounds__(256) void gemm_bf16(const __bf16* __restrict__ A,
                                                 const __bf16* __restrict__ Wt,
                                                 const float* __restrict__ bias,
                                                 void* __restrict__ out,
                                                 int M, int N, int K) {
  __shared__ __bf16 As[2][128][40];  // row-major, stride 40 (80B, 16B aligned)
  __shared__ __bf16 Bs[2][128][40];  // Bs[n][k] (Wt already [N][K])

  const int tid  = threadIdx.x;
  const int lane = tid & 31;
  const int wave = tid >> 5;
  const int wy = wave >> 1;  // 0..3 -> 32-row group
  const int wx = wave & 1;   // 0..1 -> 64-col group
  const int m0 = blockIdx.x * 128;
  const int n0 = blockIdx.y * 128;

  const int lrow  = lane & 15;
  const int akb   = (lane < 16) ? 0 : 8;   // A-frag K base (16-bit A layout)
  const int bkb   = (lane < 16) ? 0 : 8;
  const int half8 = (lane < 16) ? 0 : 8;

  f32x8 acc[2][4];
#pragma unroll
  for (int rt = 0; rt < 2; ++rt)
#pragma unroll
    for (int ct = 0; ct < 4; ++ct) acc[rt][ct] = zero8();

  // staging pointers: each thread owns two b128 chunks of A and of B per tile
  const int r = tid >> 2;            // 0..63
  const int c = (tid & 3) * 8;       // 0,8,16,24
  const __bf16* Ap0 = &A[(size_t)(m0 + r) * K + c];
  const __bf16* Ap1 = &A[(size_t)(m0 + r + 64) * K + c];
  const __bf16* Bp0 = &Wt[(size_t)(n0 + r) * K + c];
  const __bf16* Bp1 = &Wt[(size_t)(n0 + r + 64) * K + c];

  // prologue: stage tile 0
  *(bf16x8*)&As[0][r][c]      = *(const bf16x8*)Ap0;
  *(bf16x8*)&As[0][r + 64][c] = *(const bf16x8*)Ap1;
  *(bf16x8*)&Bs[0][r][c]      = *(const bf16x8*)Bp0;
  *(bf16x8*)&Bs[0][r + 64][c] = *(const bf16x8*)Bp1;
  __syncthreads();

  const int ntiles = K >> 5;
  for (int kt = 0; kt < ntiles; ++kt) {
    const int cur = kt & 1;
    const bool more = (kt + 1 < ntiles);
    bf16x8 a0, a1, b0, b1;
    if (more) {  // issue next tile's global loads before the WMMA block
      const size_t off = (size_t)(kt + 1) << 5;
      a0 = *(const bf16x8*)(Ap0 + off);
      a1 = *(const bf16x8*)(Ap1 + off);
      b0 = *(const bf16x8*)(Bp0 + off);
      b1 = *(const bf16x8*)(Bp1 + off);
    }

    bf16x16 af[2], bfr[4];
#pragma unroll
    for (int rt = 0; rt < 2; ++rt) {
      const __bf16* p = &As[cur][wy * 32 + rt * 16 + lrow][0];
      af[rt] = ld_frag(p + akb, p + akb + 16);
    }
#pragma unroll
    for (int ct = 0; ct < 4; ++ct) {
      const __bf16* p = &Bs[cur][wx * 64 + ct * 16 + lrow][0];
      bfr[ct] = ld_frag(p + bkb, p + bkb + 16);
    }
#pragma unroll
    for (int rt = 0; rt < 2; ++rt)
#pragma unroll
      for (int ct = 0; ct < 4; ++ct)
        acc[rt][ct] = wmma_bf16(af[rt], bfr[ct], acc[rt][ct]);

    if (more) {  // store next tile to the alternate buffer
      *(bf16x8*)&As[cur ^ 1][r][c]      = a0;
      *(bf16x8*)&As[cur ^ 1][r + 64][c] = a1;
      *(bf16x8*)&Bs[cur ^ 1][r][c]      = b0;
      *(bf16x8*)&Bs[cur ^ 1][r + 64][c] = b1;
    }
    __syncthreads();
  }

  // ---- epilogue: bias + store (straight-line per MODE) ----
#pragma unroll
  for (int rt = 0; rt < 2; ++rt) {
    const int mrow0 = m0 + wy * 32 + rt * 16 + half8;  // 8 consecutive rows from here
#pragma unroll
    for (int ct = 0; ct < 4; ++ct) {
      const int n = n0 + wx * 64 + ct * 16 + lrow;
      const float bv = bias[n];
      if constexpr (MODE == 0) {
        const int h = n >> 6, d = n & 63;
        const int b = mrow0 >> 11, t0 = mrow0 & (SEQ - 1);
        __bf16* dst = (__bf16*)out + ((size_t)(b * NHEAD + h) * SEQ + t0) * HDIM + d;
#pragma unroll
        for (int j = 0; j < 8; ++j) dst[j * HDIM] = (__bf16)(acc[rt][ct][j] + bv);
      } else if constexpr (MODE == 1) {
        float* dst = (float*)out + (size_t)mrow0 * N + n;
#pragma unroll
        for (int j = 0; j < 8; ++j) dst[(size_t)j * N] = acc[rt][ct][j] + bv;
      } else {  // MODE 2: V transposed [B,H,hd,T]; 8 consecutive t -> one b128
        const int h = n >> 6, d = n & 63;
        const int b = mrow0 >> 11, t0 = mrow0 & (SEQ - 1);
        bf16x8 pk;
#pragma unroll
        for (int j = 0; j < 8; ++j) pk[j] = (__bf16)(acc[rt][ct][j] + bv);
        *(bf16x8*)((__bf16*)out + ((size_t)(b * NHEAD + h) * HDIM + d) * SEQ + t0) = pk;
      }
    }
  }
}

// ---------------------------------------------------------------------------
// Causal flash attention.  Q,K bf16 [B,H,T,hd]; VT bf16 [B,H,hd,T];
// O bf16 [B,T,H*hd].  Grid: (T/64, B*H); block: 128 = 4 waves x 16 q rows.
// ---------------------------------------------------------------------------
__global__ __launch_bounds__(128) void attn_flash(const __bf16* __restrict__ Q,
                                                  const __bf16* __restrict__ Kk,
                                                  const __bf16* __restrict__ VT,
                                                  __bf16* __restrict__ O) {
  __shared__ __bf16 Ks[64][72];       // K tile: row = key, col = d
  __shared__ __bf16 Vt[64][72];       // V tile: row = d, col = key (from VT)
  __shared__ __bf16 Pw[4][16][72];    // per-wave P staging (D-layout -> A-layout)

  const int tid  = threadIdx.x;
  const int lane = tid & 31;
  const int w    = tid >> 5;          // wave 0..3
  const int bh   = blockIdx.y;        // b*16 + h
  const int q0   = blockIdx.x * 64;
  const size_t base = (size_t)bh * SEQ * HDIM;  // same constant for both layouts

  const int lrow  = lane & 15;
  const int akb   = (lane < 16) ? 0 : 8;
  const int bkb   = (lane < 16) ? 0 : 8;
  const int half8 = (lane < 16) ? 0 : 8;

  // Q fragments held in registers (rows q0 + w*16 + lrow)
  const __bf16* qrow = Q + base + (size_t)(q0 + w * 16 + lrow) * HDIM;
  bf16x16 qf0 = ld_frag(qrow + akb,      qrow + akb + 16);       // d 0..31
  bf16x16 qf1 = ld_frag(qrow + 32 + akb, qrow + 32 + akb + 16);  // d 32..63

  f32x8 o[4];
#pragma unroll
  for (int ct = 0; ct < 4; ++ct) o[ct] = zero8();
  float mstat[8], lstat[8];
#pragma unroll
  for (int j = 0; j < 8; ++j) { mstat[j] = -3.0e38f; lstat[j] = 0.f; }

  const int nblocks = (q0 >> 6) + 1;  // causal: only key blocks j0 <= q0
  for (int jb = 0; jb < nblocks; ++jb) {
    const int j0 = jb * 64;
    __syncthreads();
    // stage K and V tiles (all b128: VT is already transposed in HBM)
#pragma unroll
    for (int p = 0; p < 4; ++p) {
      int qd = tid + 128 * p;      // 0..511 chunks of 8
      int rr = qd >> 3;            // 0..63 (key row for K, d row for V)
      int c8 = (qd & 7) * 8;       // 0..56
      *(bf16x8*)&Ks[rr][c8] = *(const bf16x8*)&Kk[base + (size_t)(j0 + rr) * HDIM + c8];
      *(bf16x8*)&Vt[rr][c8] = *(const bf16x8*)&VT[base + (size_t)rr * SEQ + j0 + c8];
    }
    __syncthreads();

    // S = (Q K^T) * scale  (matmul K-dim = head dim = 64)
    f32x8 s[4];
#pragma unroll
    for (int ct = 0; ct < 4; ++ct) {
      s[ct] = zero8();
      const __bf16* kr = &Ks[ct * 16 + lrow][0];  // lane col = key
      bf16x16 b0 = ld_frag(kr + bkb,      kr + bkb + 16);       // d 0..31
      bf16x16 b1 = ld_frag(kr + 32 + bkb, kr + 32 + bkb + 16);  // d 32..63
      s[ct] = wmma_bf16(qf0, b0, s[ct]);
      s[ct] = wmma_bf16(qf1, b1, s[ct]);
    }

    const bool diag = (j0 == q0);
#pragma unroll
    for (int ct = 0; ct < 4; ++ct) {
#pragma unroll
      for (int j = 0; j < 8; ++j) {
        float v = s[ct][j] * 0.125f;  // 1/sqrt(64)
        if (diag) {
          int key = ct * 16 + lrow;
          int row = w * 16 + j + half8;
          if (key > row) v = -1.0e9f;  // matches reference NEG fill
        }
        s[ct][j] = v;
      }
    }

    // flash softmax update (lanes 0-15 hold rows j, lanes 16-31 rows j+8)
    float alpha[8];
#pragma unroll
    for (int j = 0; j < 8; ++j) {
      float rm = fmaxf(fmaxf(s[0][j], s[1][j]), fmaxf(s[2][j], s[3][j]));
      rm = fmaxf(rm, __shfl_xor(rm, 1));
      rm = fmaxf(rm, __shfl_xor(rm, 2));
      rm = fmaxf(rm, __shfl_xor(rm, 4));
      rm = fmaxf(rm, __shfl_xor(rm, 8));
      float mn = fmaxf(mstat[j], rm);
      alpha[j] = __expf(mstat[j] - mn);
      mstat[j] = mn;
    }
#pragma unroll
    for (int ct = 0; ct < 4; ++ct)
#pragma unroll
      for (int j = 0; j < 8; ++j) s[ct][j] = __expf(s[ct][j] - mstat[j]);
#pragma unroll
    for (int j = 0; j < 8; ++j) {
      float rs = s[0][j] + s[1][j] + s[2][j] + s[3][j];
      rs += __shfl_xor(rs, 1);
      rs += __shfl_xor(rs, 2);
      rs += __shfl_xor(rs, 4);
      rs += __shfl_xor(rs, 8);
      lstat[j] = lstat[j] * alpha[j] + rs;
#pragma unroll
      for (int ct = 0; ct < 4; ++ct) o[ct][j] *= alpha[j];
    }

    // P: D-layout -> bf16 A-fragments via wave-private LDS
#pragma unroll
    for (int ct = 0; ct < 4; ++ct)
#pragma unroll
      for (int j = 0; j < 8; ++j)
        Pw[w][j + half8][ct * 16 + lrow] = (__bf16)s[ct][j];
    __syncthreads();
    const __bf16* pr = &Pw[w][lrow][0];
    bf16x16 pf0 = ld_frag(pr + akb,      pr + akb + 16);       // keys 0..31
    bf16x16 pf1 = ld_frag(pr + 32 + akb, pr + 32 + akb + 16);  // keys 32..63

    // O += P @ V  (matmul K-dim = keys = 64)
#pragma unroll
    for (int ct = 0; ct < 4; ++ct) {
      const __bf16* vr = &Vt[ct * 16 + lrow][0];  // lane col = output dim
      bf16x16 vb0 = ld_frag(vr + bkb,      vr + bkb + 16);       // keys 0..31
      bf16x16 vb1 = ld_frag(vr + 32 + bkb, vr + 32 + bkb + 16);  // keys 32..63
      o[ct] = wmma_bf16(pf0, vb0, o[ct]);
      o[ct] = wmma_bf16(pf1, vb1, o[ct]);
    }
  }

  // normalize and store O as [B, T, H*hd] bf16
  const int b = bh >> 4, h = bh & 15;
#pragma unroll
  for (int ct = 0; ct < 4; ++ct) {
#pragma unroll
    for (int j = 0; j < 8; ++j) {
      int t = q0 + w * 16 + j + half8;
      float v = o[ct][j] / lstat[j];
      O[(size_t)(b * SEQ + t) * DMODEL + h * HDIM + ct * 16 + lrow] = (__bf16)v;
    }
  }
}

// ---------------------------------------------------------------------------
extern "C" void kernel_launch(void* const* d_in, const int* in_sizes, int n_in,
                              void* d_out, int out_size, void* d_ws, size_t ws_size,
                              hipStream_t stream) {
  const float* x  = (const float*)d_in[0];
  const float* Wq = (const float*)d_in[1]; const float* bq = (const float*)d_in[2];
  const float* Wk = (const float*)d_in[3]; const float* bk = (const float*)d_in[4];
  const float* Wv = (const float*)d_in[5]; const float* bv = (const float*)d_in[6];
  const float* Wo = (const float*)d_in[7]; const float* bo = (const float*)d_in[8];

  const size_t NX = (size_t)MROWS * DMODEL;   // 8388608
  const size_t NW = (size_t)DMODEL * DMODEL;  // 1048576

  char* ws = (char*)d_ws;
  __bf16* xb  = (__bf16*)ws;                 ws += NX * 2;
  __bf16* Wqt = (__bf16*)ws;                 ws += NW * 2;
  __bf16* Wkt = (__bf16*)ws;                 ws += NW * 2;
  __bf16* Wvt = (__bf16*)ws;                 ws += NW * 2;
  __bf16* Wot = (__bf16*)ws;                 ws += NW * 2;
  __bf16* Qb  = (__bf16*)ws;                 ws += NX * 2;
  __bf16* Kb  = (__bf16*)ws;                 ws += NX * 2;
  __bf16* Vb  = (__bf16*)ws;                 ws += NX * 2;   // transposed [B,H,hd,T]
  __bf16* Ob  = (__bf16*)ws;                 ws += NX * 2;

  // 1) fp32 -> bf16 (x as-is; weights transposed to [N][K])
  cvt_f32_bf16<<<(int)(NX / 4 + 255) / 256, 256, 0, stream>>>(x, xb, (int)(NX / 4));
  dim3 gt(DMODEL / 32, DMODEL / 32);
  cvt_transpose_bf16<<<gt, 256, 0, stream>>>(Wq, Wqt, DMODEL, DMODEL);
  cvt_transpose_bf16<<<gt, 256, 0, stream>>>(Wk, Wkt, DMODEL, DMODEL);
  cvt_transpose_bf16<<<gt, 256, 0, stream>>>(Wv, Wvt, DMODEL, DMODEL);
  cvt_transpose_bf16<<<gt, 256, 0, stream>>>(Wo, Wot, DMODEL, DMODEL);

  // 2) QKV projections
  dim3 gg(MROWS / 128, DMODEL / 128);
  gemm_bf16<0><<<gg, 256, 0, stream>>>(xb, Wqt, bq, Qb, MROWS, DMODEL, DMODEL);
  gemm_bf16<0><<<gg, 256, 0, stream>>>(xb, Wkt, bk, Kb, MROWS, DMODEL, DMODEL);
  gemm_bf16<2><<<gg, 256, 0, stream>>>(xb, Wvt, bv, Vb, MROWS, DMODEL, DMODEL);

  // 3) causal flash attention -> Ob [B,T,D] bf16
  dim3 ga(SEQ / 64, BATCH * NHEAD);
  attn_flash<<<ga, 128, 0, stream>>>(Qb, Kb, Vb, Ob);

  // 4) output projection (fp32 out + bias) -> d_out
  gemm_bf16<1><<<gg, 256, 0, stream>>>(Ob, Wot, bo, d_out, MROWS, DMODEL, DMODEL);
}
